// VisionTransformerPredictorAC_2156073582928
// MI455X (gfx1250) — matrix operand order repaired and checked by reading, not compiled, not run on hardware
//
#include <hip/hip_runtime.h>
#include <hip/hip_bf16.h>
#include <math.h>

#define BATCH   2
#define TFRM    4
#define PD      1024
#define EMB     768
#define NHEAD   16
#define HDIM    64
#define NDEPTH  6
#define TPFRAME 258
#define NTOK    1032                 // TFRM * TPFRAME
#define MTOK    (BATCH * NTOK)       // 2064
#define KEYPAD  1056                 // 33 * 32 (keys padded for 32-wide tiles)
#define SLACKROWS (KEYPAD - NTOK)    // 24 padded token rows in qkv buffer
#define NEGINF  (-1e30f)

typedef __attribute__((ext_vector_type(16))) _Float16 v16h;
typedef __attribute__((ext_vector_type(8)))  float    v8f;
typedef __attribute__((ext_vector_type(2)))  _Float16 h2_t;
typedef unsigned int u32x4 __attribute__((ext_vector_type(4)));
typedef int          i32x8 __attribute__((ext_vector_type(8)));
typedef int          i32x4 __attribute__((ext_vector_type(4)));

#if defined(__has_builtin)
#  if __has_builtin(__builtin_amdgcn_tensor_load_to_lds) && __has_builtin(__builtin_amdgcn_s_wait_tensorcnt)
#    define HAVE_TDM 1
#  else
#    define HAVE_TDM 0
#  endif
#else
#  define HAVE_TDM 0
#endif

union Frag { v16h v; h2_t h2[8]; float4 f4[2]; };

static __device__ __forceinline__ h2_t pk2(float a, float b) {
    h2_t r; r.x = (_Float16)a; r.y = (_Float16)b; return r;
}

static __device__ __forceinline__ v8f wmma16(v16h a, v16h b, v8f c) {
    // D = A(16x32 f16) * B(32x16 f16) + C(16x16 f32)
    return __builtin_amdgcn_wmma_f32_16x16x32_f16(false, a, false, b, (short)0, c, false, false);
}

// Convert 16 contiguous fp32 (from 4 float4s) into an f16 fragment register
static __device__ __forceinline__ void cvt16(Frag& f, float4 q0, float4 q1, float4 q2, float4 q3) {
    f.h2[0] = pk2(q0.x, q0.y); f.h2[1] = pk2(q0.z, q0.w);
    f.h2[2] = pk2(q1.x, q1.y); f.h2[3] = pk2(q1.z, q1.w);
    f.h2[4] = pk2(q2.x, q2.y); f.h2[5] = pk2(q2.z, q2.w);
    f.h2[6] = pk2(q3.x, q3.y); f.h2[7] = pk2(q3.z, q3.w);
}

// ---------------------------------------------------------------------------
// utility kernels
// ---------------------------------------------------------------------------
__global__ void zero_kernel(float* __restrict__ p, int n) {
    int i = blockIdx.x * 256 + threadIdx.x;
    if (i < n) p[i] = 0.0f;
}

// Pack fp32 weight (K x N row-major) into fragment-contiguous f16 blocks:
//   element (kt, n, hi, j) at [((kt*N + n)*2 + hi)*8 + j] = {W[kt*32+(j<4 ? hi*8+2j : 16+hi*8+2(j-4))][n], W[..+1][n]}
// so one lane's whole B fragment (32 bytes) is contiguous.
__global__ void pack_kernel(const float* __restrict__ w, h2_t* __restrict__ wp, int K, int N) {
    int idx = blockIdx.x * 256 + threadIdx.x;
    int total = (K >> 1) * N;
    if (idx >= total) return;
    int j = idx & 7;
    int rest = idx >> 3;
    int hi = rest & 1; rest >>= 1;
    int n = rest % N, kt = rest / N;
    int k = kt * 32 + (j < 4 ? hi * 8 + 2 * j : 16 + hi * 8 + 2 * (j - 4));
    wp[idx] = pk2(w[(size_t)k * N + n], w[(size_t)(k + 1) * N + n]);
}

// ---------------------------------------------------------------------------
// WMMA GEMM:  C[M,N] = act(A[M,K] * W[K,N] + bias) (+ resid)
// block = 128 (4 waves); each wave computes a 32x64 tile (2 m-tiles x 4 n-tiles),
// so every B fragment fetch (2 x b128) feeds 2 WMMAs.
// ---------------------------------------------------------------------------
__global__ __launch_bounds__(128)
void gemm_kernel(const float* __restrict__ A, const h2_t* __restrict__ Wp,
                 const float* __restrict__ bias, float* __restrict__ C,
                 const float* __restrict__ resid, int M, int N, int K, int act) {
    int wave = threadIdx.x >> 5;
    int lane = threadIdx.x & 31;
    int mt2 = blockIdx.x * 4 + wave;           // 32-row tile index
    if (mt2 * 32 >= M) return;
    int nb = blockIdx.y;
    int l15 = lane & 15, hi = lane >> 4;
    int kbA = hi * 8;
    int colb = nb * 64 + l15;
    int row0 = mt2 * 32 + l15;
    int row1 = row0 + 16;
    bool has2 = (mt2 * 32 + 16) < M;
    if (!has2) row1 = row0;                    // clamp reads; stores guarded

    v8f acc[2][4];
    #pragma unroll
    for (int m = 0; m < 2; ++m)
        #pragma unroll
        for (int t = 0; t < 4; ++t) acc[m][t] = (v8f)0.0f;

    const float* ap0 = A + (size_t)row0 * K;
    const float* ap1 = A + (size_t)row1 * K;
    const h2_t* wb = Wp + ((size_t)colb * 2 + hi) * 8;   // + kt*N*16 per k-tile
    int nkt = K >> 5;

    for (int kt = 0; kt < nkt; ++kt) {
        int k0 = kt * 32;
        Frag a0, a1;
        cvt16(a0, *(const float4*)(ap0 + k0 + kbA),      *(const float4*)(ap0 + k0 + kbA + 4),
                  *(const float4*)(ap0 + k0 + 16 + kbA), *(const float4*)(ap0 + k0 + 20 + kbA));
        cvt16(a1, *(const float4*)(ap1 + k0 + kbA),      *(const float4*)(ap1 + k0 + kbA + 4),
                  *(const float4*)(ap1 + k0 + 16 + kbA), *(const float4*)(ap1 + k0 + 20 + kbA));
        const h2_t* wkt = wb + (size_t)kt * N * 16;
        if (kt + 1 < nkt)
            __builtin_prefetch(wkt + (size_t)N * 16, 0, 1);   // next K-panel
        #pragma unroll
        for (int t = 0; t < 4; ++t) {
            Frag bf;
            const h2_t* p = wkt + t * 256;     // (t*16 cols)*2*8 half2
            bf.f4[0] = *(const float4*)p;
            bf.f4[1] = *(const float4*)(p + 4);
            acc[0][t] = wmma16(a0.v, bf.v, acc[0][t]);
            acc[1][t] = wmma16(a1.v, bf.v, acc[1][t]);
        }
    }

    // epilogue: C layout — lane l15 = col, VGPR r = row (r + 8*hi)
    #pragma unroll
    for (int m = 0; m < 2; ++m) {
        if (m == 1 && !has2) break;
        int crow = mt2 * 32 + m * 16 + hi * 8;
        #pragma unroll
        for (int t = 0; t < 4; ++t) {
            int col = colb + t * 16;
            float bv = bias ? bias[col] : 0.0f;
            #pragma unroll
            for (int r = 0; r < 8; ++r) {
                float v = acc[m][t][r] + bv;
                if (act == 1) v = 0.5f * v * (1.0f + erff(v * 0.70710678118654752f));
                size_t ci = (size_t)(crow + r) * N + col;
                if (resid) v += resid[ci];
                C[ci] = v;
            }
        }
    }
}

// ---------------------------------------------------------------------------
// LayerNorm over 1024 dims; one row per block (256 threads)
// ---------------------------------------------------------------------------
__global__ __launch_bounds__(256)
void ln_kernel(const float* __restrict__ in, const float* __restrict__ g,
               const float* __restrict__ be, float* __restrict__ out) {
    int row = blockIdx.x, tid = threadIdx.x;
    const float* x = in + (size_t)row * PD;
    float s = 0.f, s2 = 0.f;
    for (int d = tid; d < PD; d += 256) { float v = x[d]; s += v; s2 += v * v; }
    __shared__ float sb[256], sb2[256];
    sb[tid] = s; sb2[tid] = s2; __syncthreads();
    for (int st = 128; st > 0; st >>= 1) {
        if (tid < st) { sb[tid] += sb[tid + st]; sb2[tid] += sb2[tid + st]; }
        __syncthreads();
    }
    float mean = sb[0] * (1.0f / PD);
    float var  = sb2[0] * (1.0f / PD) - mean * mean;
    float inv  = rsqrtf(var + 1e-5f);
    float* o = out + (size_t)row * PD;
    for (int d = tid; d < PD; d += 256) o[d] = (x[d] - mean) * inv * g[d] + be[d];
}

// ---------------------------------------------------------------------------
// Assemble token stream: [action, state, 256 patches] per frame
// ---------------------------------------------------------------------------
__global__ __launch_bounds__(256)
void assemble_kernel(const float* __restrict__ xe, const float* __restrict__ actions,
                     const float* __restrict__ states, const float* __restrict__ ae_w,
                     const float* __restrict__ ae_b, const float* __restrict__ se_w,
                     const float* __restrict__ se_b, float* __restrict__ h) {
    int rowg = blockIdx.x;                 // 0..MTOK-1
    int b = rowg / NTOK, i = rowg - b * NTOK;
    int t = i / TPFRAME, j = i - t * TPFRAME;
    float* dst = h + (size_t)rowg * PD;
    int tid = threadIdx.x;
    if (j >= 2) {
        const float* src = xe + (size_t)(b * 1024 + t * 256 + (j - 2)) * PD;
        for (int d = tid; d < PD; d += 256) dst[d] = src[d];
    } else {
        const float* w  = (j == 0) ? ae_w : se_w;
        const float* bb = (j == 0) ? ae_b : se_b;
        const float* av = ((j == 0) ? actions : states) + (size_t)(b * TFRM + t) * 7;
        for (int d = tid; d < PD; d += 256) {
            float s = bb[d];
            #pragma unroll
            for (int q = 0; q < 7; ++q) s += av[q] * w[q * PD + d];
            dst[d] = s;
        }
    }
}

// ---------------------------------------------------------------------------
// RoPE-3D in place on q and k slices of qkv buffer
// ---------------------------------------------------------------------------
__global__ __launch_bounds__(512)
void rope_kernel(float* __restrict__ qkv) {
    int token = blockIdx.x;                 // 0..MTOK-1
    int n = token % NTOK;
    int tid = threadIdx.x;
    if (tid >= NHEAD * 30) return;          // 16 heads * 30 rotation pairs
    int head = tid / 30, pr = tid - head * 30;
    int chunk = pr / 10, i = pr - chunk * 10;
    int frame = n / TPFRAME;
    int rem = n - frame * TPFRAME;
    int patch = (rem >= 2) ? (rem - 2) : 0;
    float pos;
    if (chunk == 0)      pos = (float)frame;
    else if (chunk == 1) pos = (float)(patch / 16);
    else                 pos = (float)(patch % 16);
    float omega = __powf(10000.0f, -(float)i * 0.1f);
    float ang = pos * omega;
    float c = __cosf(ang), s = __sinf(ang);
    size_t base = (size_t)token * (3 * PD) + head * HDIM + chunk * 20 + 2 * i;
    #pragma unroll
    for (int sel = 0; sel < 2; ++sel) {     // q then k
        float* p = qkv + base + (size_t)sel * PD;
        float x1 = p[0], x2 = p[1];
        p[0] = x1 * c - x2 * s;
        p[1] = x2 * c + x1 * s;
    }
}

// ---------------------------------------------------------------------------
// V transpose: vt[b,h][d][key] = V[b,key][h,d], padded keys -> zeroed slack.
// 32x32 fp32 tile staged in LDS. On CDNA5 the strided 2D tile is pulled with
// the Tensor Data Mover (TENSOR_LOAD_TO_LDS, pad 1 dword / 32 dwords -> the
// conflict-free 33-stride tile); fallback = cooperative b32 loads.
// block (32,8), grid (KEYPAD/32, HDIM/32, BATCH*NHEAD)
// ---------------------------------------------------------------------------
__global__ __launch_bounds__(256)
void transv_kernel(const float* __restrict__ qkv, float* __restrict__ vt) {
    __shared__ float tile[32][33];
    int kt = blockIdx.x, dchunk = blockIdx.y, bh = blockIdx.z;
    int b = bh >> 4, hh = bh & 15;
    int key0 = kt * 32, d0 = dchunk * 32;
    int tx = threadIdx.x, ty = threadIdx.y;
    const float* src = qkv + (size_t)(b * NTOK + key0) * (3 * PD) + 2 * PD + hh * HDIM + d0;
#if HAVE_TDM
    if (ty == 0) {   // TDM issues per-wave (EXEC ignored) -> only wave 0 issues
        unsigned ldsOff = (unsigned)(size_t)&tile[0][0];
        unsigned long long ga = (unsigned long long)(size_t)src;
        u32x4 g0;
        g0.x = 1u;                                   // count=1, user descriptor
        g0.y = ldsOff;                               // lds_addr (bytes)
        g0.z = (unsigned)ga;                         // global_addr[31:0]
        g0.w = (unsigned)((ga >> 32) & 0x1ffffffu) | (2u << 30);  // addr[56:32], type=2
        i32x8 g1;
        g1[0] = (int)((2u << 16) | (1u << 20) | (4u << 22));  // 4B elems, pad 1dw/32dw
        g1[1] = (int)(32u << 16);                    // tensor_dim0 = 32
        g1[2] = (int)(32u << 16);                    // tensor_dim1 = 32
        g1[3] = (int)(32u << 16);                    // tile_dim0 = 32
        g1[4] = 32;                                  // tile_dim1 = 32
        g1[5] = 3 * PD;                              // tensor_dim0_stride = 3072
        g1[6] = 0; g1[7] = 0;
        i32x4 z4 = (i32x4)0;
#  if __clang_major__ >= 23
        i32x8 z8 = (i32x8)0;
        __builtin_amdgcn_tensor_load_to_lds(g0, g1, z4, z4, z8, 0);
#  else
        __builtin_amdgcn_tensor_load_to_lds(g0, g1, z4, z4, 0);
#  endif
        __builtin_amdgcn_s_wait_tensorcnt(0);
    }
    __syncthreads();
#else
    #pragma unroll
    for (int i = 0; i < 4; ++i) {
        int kl = ty + i * 8;
        tile[kl][tx] = src[(size_t)kl * (3 * PD) + tx];   // padded keys read zeroed slack
    }
    __syncthreads();
#endif
    float* dst = vt + ((size_t)bh * HDIM + d0) * KEYPAD + key0;
    #pragma unroll
    for (int i = 0; i < 4; ++i) {
        int dl = ty + i * 8;
        dst[(size_t)dl * KEYPAD + tx] = tile[tx][dl];
    }
}

// ---------------------------------------------------------------------------
// Flash attention: one wave per (qtile, head, batch). Key tiles of 32 (padded
// to KEYPAD). Frame-causal mask. V read from transposed vt (coalesced b128).
// ---------------------------------------------------------------------------
__global__ __launch_bounds__(32)
void attn_kernel(const float* __restrict__ qkv, const float* __restrict__ vt,
                 float* __restrict__ o) {
    __shared__ _Float16 smemP[16 * 32];
    int qt = blockIdx.x, hh = blockIdx.y, b = blockIdx.z;
    int lane = threadIdx.x;
    int l15 = lane & 15, hi = lane >> 4;
    int kbA = hi * 16, kbB = hi * 16;

    // Q fragments (pre-scaled by 1/sqrt(HDIM)); rows >= NTOK read zeroed slack
    int qrow = qt * 16 + l15;
    const float* qp = qkv + (size_t)(b * NTOK + qrow) * (3 * PD) + hh * HDIM;
    Frag qa[2];
    #pragma unroll
    for (int c = 0; c < 2; ++c) {
        float4 q0 = *(const float4*)(qp + c * 32 + kbA);
        float4 q1 = *(const float4*)(qp + c * 32 + kbA + 4);
        float4 q2 = *(const float4*)(qp + c * 32 + kbA + 8);
        float4 q3 = *(const float4*)(qp + c * 32 + kbA + 12);
        qa[c].h2[0] = pk2(0.125f * q0.x, 0.125f * q0.y);
        qa[c].h2[1] = pk2(0.125f * q0.z, 0.125f * q0.w);
        qa[c].h2[2] = pk2(0.125f * q1.x, 0.125f * q1.y);
        qa[c].h2[3] = pk2(0.125f * q1.z, 0.125f * q1.w);
        qa[c].h2[4] = pk2(0.125f * q2.x, 0.125f * q2.y);
        qa[c].h2[5] = pk2(0.125f * q2.z, 0.125f * q2.w);
        qa[c].h2[6] = pk2(0.125f * q3.x, 0.125f * q3.y);
        qa[c].h2[7] = pk2(0.125f * q3.z, 0.125f * q3.w);
    }

    float mrow[8], lrow[8];
    int fq[8];
    v8f oa[4];
    #pragma unroll
    for (int r = 0; r < 8; ++r) {
        mrow[r] = NEGINF; lrow[r] = 0.0f;
        fq[r] = (qt * 16 + r + 8 * hi) / TPFRAME;
    }
    #pragma unroll
    for (int dc = 0; dc < 4; ++dc) oa[dc] = (v8f)0.0f;

    const float* vtb = vt + (size_t)(b * NHEAD + hh) * HDIM * KEYPAD;

    for (int kt = 0; kt < KEYPAD / 32; ++kt) {
        int key0 = kt * 32;
        v8f s0 = (v8f)0.0f, s1 = (v8f)0.0f;
        // logits = Q * K^T  (B operand: cols = keys, K-dim = d)
        #pragma unroll
        for (int t2 = 0; t2 < 2; ++t2) {
            const float* kp = qkv + (size_t)(b * NTOK + key0 + t2 * 16 + l15) * (3 * PD)
                              + PD + hh * HDIM + kbB;
            Frag k0f, k1f;
            cvt16(k0f, *(const float4*)(kp),      *(const float4*)(kp + 4),
                       *(const float4*)(kp + 8),  *(const float4*)(kp + 12));
            cvt16(k1f, *(const float4*)(kp + 32), *(const float4*)(kp + 36),
                       *(const float4*)(kp + 40), *(const float4*)(kp + 44));
            if (t2 == 0) { s0 = wmma16(qa[0].v, k0f.v, s0); s0 = wmma16(qa[1].v, k1f.v, s0); }
            else         { s1 = wmma16(qa[0].v, k0f.v, s1); s1 = wmma16(qa[1].v, k1f.v, s1); }
        }
        // mask + online softmax (C layout: col = l15, row = r + 8*hi)
        int c0 = key0 + l15, c1 = key0 + 16 + l15;
        int fk0 = c0 / TPFRAME, fk1 = c1 / TPFRAME;
        float pv0[8], pv1[8];
        #pragma unroll
        for (int r = 0; r < 8; ++r) {
            float v0 = ((c0 < NTOK) && (fq[r] >= fk0)) ? s0[r] : NEGINF;
            float v1 = ((c1 < NTOK) && (fq[r] >= fk1)) ? s1[r] : NEGINF;
            float rm = fmaxf(v0, v1);
            rm = fmaxf(rm, __shfl_xor(rm, 1, 32));
            rm = fmaxf(rm, __shfl_xor(rm, 2, 32));
            rm = fmaxf(rm, __shfl_xor(rm, 4, 32));
            rm = fmaxf(rm, __shfl_xor(rm, 8, 32));
            float mnew = fmaxf(mrow[r], rm);
            float sc = __expf(mrow[r] - mnew);
            float p0 = __expf(v0 - mnew), p1 = __expf(v1 - mnew);
            float ps = p0 + p1;
            ps += __shfl_xor(ps, 1, 32);
            ps += __shfl_xor(ps, 2, 32);
            ps += __shfl_xor(ps, 4, 32);
            ps += __shfl_xor(ps, 8, 32);
            lrow[r] = lrow[r] * sc + ps;
            mrow[r] = mnew;
            oa[0][r] *= sc; oa[1][r] *= sc; oa[2][r] *= sc; oa[3][r] *= sc;
            pv0[r] = p0; pv1[r] = p1;
        }
        // stage P through LDS to convert C-fragment layout -> A-fragment layout
        __syncthreads();
        #pragma unroll
        for (int r = 0; r < 8; ++r) {
            smemP[(r + 8 * hi) * 32 + l15]      = (_Float16)pv0[r];
            smemP[(r + 8 * hi) * 32 + 16 + l15] = (_Float16)pv1[r];
        }
        __syncthreads();
        Frag pa;
        #pragma unroll
        for (int j = 0; j < 4; ++j) {
            pa.h2[j]     = *(const h2_t*)&smemP[l15 * 32 + kbA + 2 * j];
            pa.h2[j + 4] = *(const h2_t*)&smemP[l15 * 32 + kbA + 8 + 2 * j];
        }
        // O += P * V  (B operand: cols = d-chunk, K-dim = keys; vt is key-major)
        #pragma unroll
        for (int dc = 0; dc < 4; ++dc) {
            Frag vb;
            const float* vp = vtb + (size_t)(dc * 16 + l15) * KEYPAD + key0 + kbB;
            cvt16(vb, *(const float4*)(vp),     *(const float4*)(vp + 4),
                      *(const float4*)(vp + 8), *(const float4*)(vp + 12));
            oa[dc] = wmma16(pa.v, vb.v, oa[dc]);
        }
    }
    // normalize and write
    #pragma unroll
    for (int r = 0; r < 8; ++r) {
        int row = qt * 16 + r + 8 * hi;
        if (row < NTOK) {
            float inv = 1.0f / lrow[r];
            #pragma unroll
            for (int dc = 0; dc < 4; ++dc)
                o[(size_t)(b * NTOK + row) * PD + hh * HDIM + dc * 16 + l15] = oa[dc][r] * inv;
        }
    }
}

// ---------------------------------------------------------------------------
// Gather patch tokens (drop action/state tokens)
// ---------------------------------------------------------------------------
__global__ __launch_bounds__(256)
void gather_kernel(const float* __restrict__ h, float* __restrict__ out) {
    int row = blockIdx.x;                   // 0..BATCH*1024-1
    int b = row / 1024, p = row - b * 1024;
    int t = p / 256, q = p - t * 256;
    const float* src = h + (size_t)(b * NTOK + t * TPFRAME + 2 + q) * PD;
    float* dst = out + (size_t)row * PD;
    for (int d = threadIdx.x; d < PD; d += 256) dst[d] = src[d];
}

// ---------------------------------------------------------------------------
// host launcher
// ---------------------------------------------------------------------------
extern "C" void kernel_launch(void* const* d_in, const int* in_sizes, int n_in,
                              void* d_out, int out_size, void* d_ws, size_t ws_size,
                              hipStream_t stream) {
    const float* x       = (const float*)d_in[0];
    const float* actions = (const float*)d_in[1];
    const float* states  = (const float*)d_in[2];
    const float* pe_w    = (const float*)d_in[3];
    const float* pe_b    = (const float*)d_in[4];
    const float* ae_w    = (const float*)d_in[5];
    const float* ae_b    = (const float*)d_in[6];
    const float* se_w    = (const float*)d_in[7];
    const float* se_b    = (const float*)d_in[8];
    const float* ln1_g   = (const float*)d_in[9];
    const float* ln1_b   = (const float*)d_in[10];
    const float* qkv_w   = (const float*)d_in[11];
    const float* qkv_b   = (const float*)d_in[12];
    const float* proj_w  = (const float*)d_in[13];
    const float* proj_b  = (const float*)d_in[14];
    const float* ln2_g   = (const float*)d_in[15];
    const float* ln2_b   = (const float*)d_in[16];
    const float* fc1_w   = (const float*)d_in[17];
    const float* fc1_b   = (const float*)d_in[18];
    const float* fc2_w   = (const float*)d_in[19];
    const float* fc2_b   = (const float*)d_in[20];
    const float* norm_g  = (const float*)d_in[21];
    const float* norm_b  = (const float*)d_in[22];
    const float* out_w   = (const float*)d_in[23];
    const float* out_b   = (const float*)d_in[24];

    float* ws = (float*)d_ws;
    size_t off = 0;
    float* xe   = ws + off; off += (size_t)BATCH * 1024 * PD;        // patch embeds
    float* hbuf = ws + off; off += (size_t)MTOK * PD;                // residual stream
    float* ybuf = ws + off; off += (size_t)MTOK * PD;                // LN output
    float* obuf = ws + off; off += (size_t)MTOK * PD;                // attn out / gathered
    float* qkvb = ws + off; off += (size_t)(MTOK + SLACKROWS) * 3 * PD;
    float* hid  = ws + off; off += (size_t)MTOK * 4 * PD;            // MLP hidden
    float* vt   = ws + off; off += (size_t)BATCH * NHEAD * HDIM * KEYPAD;  // V^T
    h2_t*  wpk  = (h2_t*)(ws + off);                                 // 2M half2 = 8 MB

    auto pack = [&](const float* w, int K, int N) {
        int tot = (K / 2) * N;
        pack_kernel<<<(tot + 255) / 256, 256, 0, stream>>>(w, wpk, K, N);
    };
    auto gemm = [&](const float* A, const float* bias, float* C, const float* resid,
                    int M, int N, int K, int act) {
        int mt2 = (M + 31) / 32;
        dim3 g((mt2 + 3) / 4, N / 64);
        gemm_kernel<<<g, 128, 0, stream>>>(A, wpk, bias, C, resid, M, N, K, act);
    };

    // zero padded key rows once (attention/transpose read them; p==0 keeps them inert)
    zero_kernel<<<(SLACKROWS * 3 * PD + 255) / 256, 256, 0, stream>>>(
        qkvb + (size_t)MTOK * 3 * PD, SLACKROWS * 3 * PD);

    // patch embed + token assembly
    pack(pe_w, EMB, PD);
    gemm(x, pe_b, xe, nullptr, BATCH * 1024, PD, EMB, 0);
    assemble_kernel<<<MTOK, 256, 0, stream>>>(xe, actions, states, ae_w, ae_b, se_w, se_b, hbuf);

    for (int L = 0; L < NDEPTH; ++L) {
        ln_kernel<<<MTOK, 256, 0, stream>>>(hbuf, ln1_g + L * PD, ln1_b + L * PD, ybuf);
        pack(qkv_w + (size_t)L * PD * 3 * PD, PD, 3 * PD);
        gemm(ybuf, qkv_b + L * 3 * PD, qkvb, nullptr, MTOK, 3 * PD, PD, 0);
        rope_kernel<<<MTOK, 512, 0, stream>>>(qkvb);
        transv_kernel<<<dim3(KEYPAD / 32, HDIM / 32, BATCH * NHEAD), dim3(32, 8), 0, stream>>>(qkvb, vt);
        attn_kernel<<<dim3(KEYPAD / 16, NHEAD, BATCH), 32, 0, stream>>>(qkvb, vt, obuf);
        pack(proj_w + (size_t)L * PD * PD, PD, PD);
        gemm(obuf, proj_b + L * PD, hbuf, hbuf, MTOK, PD, PD, 0);
        ln_kernel<<<MTOK, 256, 0, stream>>>(hbuf, ln2_g + L * PD, ln2_b + L * PD, ybuf);
        pack(fc1_w + (size_t)L * PD * 4 * PD, PD, 4 * PD);
        gemm(ybuf, fc1_b + L * 4 * PD, hid, nullptr, MTOK, 4 * PD, PD, 1 /*gelu*/);
        pack(fc2_w + (size_t)L * 4 * PD * PD, 4 * PD, PD);
        gemm(hid, fc2_b + L * PD, hbuf, hbuf, MTOK, PD, 4 * PD, 0);
    }

    // drop action/state tokens, final LN, output projection
    gather_kernel<<<BATCH * 1024, 256, 0, stream>>>(hbuf, obuf);
    ln_kernel<<<BATCH * 1024, 256, 0, stream>>>(obuf, norm_g, norm_b, ybuf);
    pack(out_w, PD, EMB);
    gemm(ybuf, out_b, (float*)d_out, nullptr, BATCH * 1024, EMB, PD, 0);
}
